// FocusedPerspective2BEVConverter_84335977824276
// MI455X (gfx1250) — matrix operand order repaired and checked by reading, not compiled, not run on hardware
//
#include <hip/hip_runtime.h>
#include <math.h>

// ---- problem constants ----
#define P2B_HH 32
#define P2B_WW 88
#define P2B_HW (P2B_HH * P2B_WW)        // 2816
#define P2B_NB 6                         // BS*NUM_CAM
#define P2B_C  256
#define P2B_BEVN 40000                   // 200*200
#define P2B_MINVAR 0.00015f
#define P2B_MAXVAR 0.02f
// padded spatial planes (halo 2 on all sides, zero-filled)
#define P2B_PH 36
#define P2B_PW 92
#define P2B_PHW (P2B_PH * P2B_PW)       // 3312

typedef __bf16 bf16_t;
typedef __attribute__((ext_vector_type(16))) __bf16 v16bf;
typedef __attribute__((ext_vector_type(8)))  __bf16 v8bf;
typedef __attribute__((ext_vector_type(8)))  float  v8f;

// ---------------------------------------------------------------- utilities
__global__ void k_zero_f32(float* p, long n) {
  long i = (long)blockIdx.x * blockDim.x + threadIdx.x;
  if (i < n) p[i] = 0.0f;
}

// feats (6,256,HW) f32 NCHW -> channels 0..255 of padded NHWC (6,36,92,288) bf16
__global__ void k_feats_to_in(const float* __restrict__ feats, bf16_t* __restrict__ in1) {
  long i = (long)blockIdx.x * blockDim.x + threadIdx.x;
  const long n = (long)P2B_NB * P2B_HW * P2B_C;
  if (i >= n) return;
  int c = (int)(i % P2B_C);
  long r = i / P2B_C;
  int p = (int)(r % P2B_HW);
  int b = (int)(r / P2B_HW);
  int y = p / P2B_WW, x = p % P2B_WW;
  float v = feats[((long)b * P2B_C + c) * P2B_HW + p];
  in1[((long)(b * P2B_PH + y + 2) * P2B_PW + (x + 2)) * 288 + c] = (bf16_t)v;
}

// pack conv weights (Cout,Cin,3,3) f32 into WMMA A-fragment order, bf16.
// linear: coT*(9*nch*512) + tap*(nch*512) + chunk*512 + lane*16 + e
__global__ void k_pack_w(const float* __restrict__ w, bf16_t* __restrict__ wp,
                         int Cin, int coutTiles) {
  int t = blockIdx.x * blockDim.x + threadIdx.x;
  const int nch = Cin >> 5;
  const int total = coutTiles * 9 * nch * 512;
  if (t >= total) return;
  int e     = t & 15;
  int lane  = (t >> 4) & 31;
  int chunk = (t >> 9) % nch;
  int tap   = (t / (512 * nch)) % 9;
  int coT   = t / (512 * nch * 9);
  int m  = lane & 15;
  int hi = (lane >> 4) & 1;
  int ka = e + hi * 16;     // A 16x32 bf16 K mapping
  int ci = chunk * 32 + ka;
  int co = coT * 16 + m;
  wp[t] = (bf16_t)w[((long)co * Cin + ci) * 9 + tap];
}

// ---------------------------------------------------------------- mask tower (scalar, tiny)
__global__ void k_mask1(const float* __restrict__ mask, const float* __restrict__ W1,
                        float* __restrict__ t1) {
  int i = blockIdx.x * blockDim.x + threadIdx.x;
  const int n = P2B_NB * 16 * P2B_HW;
  if (i >= n) return;
  int p = i % P2B_HW; int r = i / P2B_HW; int co = r % 16; int b = r / 16;
  int y = p / P2B_WW, x = p % P2B_WW;
  float acc = 0.0f;
  for (int ky = 0; ky < 7; ++ky) {
    int iy = y + ky - 3; if (iy < 0 || iy >= P2B_HH) continue;
    for (int kx = 0; kx < 7; ++kx) {
      int ix = x + kx - 3; if (ix < 0 || ix >= P2B_WW) continue;
      acc += (mask[b * P2B_HW + iy * P2B_WW + ix] * (1.0f / 50.0f)) * W1[co * 49 + ky * 7 + kx];
    }
  }
  t1[i] = fmaxf(acc, 0.0f);
}

__global__ void k_mask2(const float* __restrict__ t1, const float* __restrict__ W2,
                        float* __restrict__ t2) {
  int i = blockIdx.x * blockDim.x + threadIdx.x;
  const int n = P2B_NB * 32 * P2B_HW;
  if (i >= n) return;
  int p = i % P2B_HW; int r = i / P2B_HW; int co = r % 32; int b = r / 32;
  int y = p / P2B_WW, x = p % P2B_WW;
  float acc = 0.0f;
  for (int ci = 0; ci < 16; ++ci) {
    const float* ir = t1 + ((long)b * 16 + ci) * P2B_HW;
    const float* wr = W2 + ((long)co * 16 + ci) * 9;
    for (int ky = 0; ky < 3; ++ky) {
      int iy = y + ky - 1; if (iy < 0 || iy >= P2B_HH) continue;
      for (int kx = 0; kx < 3; ++kx) {
        int ix = x + kx - 1; if (ix < 0 || ix >= P2B_WW) continue;
        acc += ir[iy * P2B_WW + ix] * wr[ky * 3 + kx];
      }
    }
  }
  t2[i] = fmaxf(acc, 0.0f);
}

// m = sigmoid(conv(t2,W3)); bf16 into channels 256..287 of padded NHWC in1,in2,in3
__global__ void k_mask3(const float* __restrict__ t2, const float* __restrict__ W3,
                        bf16_t* __restrict__ in1, bf16_t* __restrict__ in2,
                        bf16_t* __restrict__ in3) {
  int i = blockIdx.x * blockDim.x + threadIdx.x;
  const int n = P2B_NB * 32 * P2B_HW;
  if (i >= n) return;
  int p = i % P2B_HW; int r = i / P2B_HW; int co = r % 32; int b = r / 32;
  int y = p / P2B_WW, x = p % P2B_WW;
  float acc = 0.0f;
  for (int ci = 0; ci < 32; ++ci) {
    const float* ir = t2 + ((long)b * 32 + ci) * P2B_HW;
    const float* wr = W3 + ((long)co * 32 + ci) * 9;
    for (int ky = 0; ky < 3; ++ky) {
      int iy = y + ky - 1; if (iy < 0 || iy >= P2B_HH) continue;
      for (int kx = 0; kx < 3; ++kx) {
        int ix = x + kx - 1; if (ix < 0 || ix >= P2B_WW) continue;
        acc += ir[iy * P2B_WW + ix] * wr[ky * 3 + kx];
      }
    }
  }
  float s = 1.0f / (1.0f + expf(-acc));
  long di = ((long)(b * P2B_PH + y + 2) * P2B_PW + (x + 2)) * 288 + 256 + co;
  bf16_t sb = (bf16_t)s;
  in1[di] = sb; in2[di] = sb; in3[di] = sb;
}

// ---------------------------------------------------------------- WMMA implicit-GEMM 3x3 conv
// Padded NHWC activations; pre-packed weights. One wave -> 16 cout x 32 pixels
// (one A fragment feeds two B fragments -> two accumulators; halves A traffic).
__global__ void k_wmma_conv3x3(const bf16_t* __restrict__ in, int inCtot, int Cin,
                               const bf16_t* __restrict__ wp,
                               bf16_t* __restrict__ out, int outCtot, int coutTiles) {
  const int lane = threadIdx.x & 31;
  const int waveInBlock = threadIdx.x >> 5;
  long waveId = (long)blockIdx.x * (blockDim.x >> 5) + waveInBlock;
  const int pixTiles2 = P2B_HW / 32;                // 88 double-tiles
  const int tilesPerImage = coutTiles * pixTiles2;
  int b = (int)(waveId / tilesPerImage);
  int rr = (int)(waveId % tilesPerImage);
  int coTile = rr / pixTiles2;
  int pTile  = rr % pixTiles2;
  if (b >= P2B_NB) return;

  const bool hi = (lane & 16) != 0;
  const int n = lane & 15;
  const int p0 = pTile * 32 + n;                    // first pixel group
  const int p1 = p0 + 16;                           // second pixel group
  const int py0 = p0 / P2B_WW, px0 = p0 % P2B_WW;
  const int py1 = p1 / P2B_WW, px1 = p1 % P2B_WW;
  const int nch = Cin >> 5;

  const bf16_t* wbase = wp + ((long)coTile * 9) * nch * 512 + lane * 16;
  v8f acc0 = {}, acc1 = {};
  for (int tap = 0; tap < 9; ++tap) {
    const int dy = tap / 3 - 1, dx = tap % 3 - 1;
    const bf16_t* ib0 = in
        + ((long)(b * P2B_PH + py0 + dy + 2) * P2B_PW + (px0 + dx + 2)) * inCtot
        + (hi ? 16 : 0);
    const bf16_t* ib1 = in
        + ((long)(b * P2B_PH + py1 + dy + 2) * P2B_PW + (px1 + dx + 2)) * inCtot
        + (hi ? 16 : 0);
    // hint L2 about next row of activations (global_prefetch_b8)
    __builtin_prefetch(ib0 + (long)P2B_PW * inCtot, 0, 1);
    const bf16_t* wt = wbase + (long)tap * nch * 512;
    for (int c = 0; c < nch; ++c) {
      v16bf afrag = *(const v16bf*)(wt + (long)c * 512);
      v16bf bf0   = *(const v16bf*)(ib0 + c * 32);
      v16bf bf1   = *(const v16bf*)(ib1 + c * 32);
      acc0 = __builtin_amdgcn_wmma_f32_16x16x32_bf16(
          false, afrag, false, bf0, (short)0, acc0, false, false);
      acc1 = __builtin_amdgcn_wmma_f32_16x16x32_bf16(
          false, afrag, false, bf1, (short)0, acc1, false, false);
    }
  }
  // C/D: VGPR r -> channel coTile*16 + r + 8*hi ; 8 consecutive channels -> one 16B store
  v8bf ov0, ov1;
#pragma unroll
  for (int r8 = 0; r8 < 8; ++r8) {
    float a0 = acc0[r8], a1 = acc1[r8];
    ov0[r8] = (bf16_t)(a0 > 0.0f ? a0 : 0.0f);
    ov1[r8] = (bf16_t)(a1 > 0.0f ? a1 : 0.0f);
  }
  const int coOff = coTile * 16 + (hi ? 8 : 0);
  long opix0 = (long)(b * P2B_PH + py0 + 2) * P2B_PW + (px0 + 2);
  long opix1 = (long)(b * P2B_PH + py1 + 2) * P2B_PW + (px1 + 2);
  *(v8bf*)(out + opix0 * outCtot + coOff) = ov0;
  *(v8bf*)(out + opix1 * outCtot + coOff) = ov1;
}

// ---------------------------------------------------------------- 5x5 sigmoid heads (scalar)
__global__ void k_head5(const bf16_t* __restrict__ in, int inCtot, int CinUsed,
                        const float* __restrict__ w, int Cout,
                        float* __restrict__ out, int mode) {
  int i = blockIdx.x * blockDim.x + threadIdx.x;
  const int n = P2B_NB * Cout * P2B_HW;
  if (i >= n) return;
  int p = i % P2B_HW; int r = i / P2B_HW; int co = r % Cout; int b = r / Cout;
  int y = p / P2B_WW, x = p % P2B_WW;
  float acc = 0.0f;
  for (int ky = 0; ky < 5; ++ky) {
    for (int kx = 0; kx < 5; ++kx) {
      const bf16_t* ir = in + ((long)(b * P2B_PH + y + ky) * P2B_PW + (x + kx)) * inCtot;
      const float*  wr = w + (long)co * CinUsed * 25 + ky * 5 + kx;
      for (int ci = 0; ci < CinUsed; ++ci)
        acc += (float)ir[ci] * wr[(long)ci * 25];
    }
  }
  float s = 1.0f / (1.0f + expf(-acc));
  if (mode == 0) {
    float scale = (co == 0) ? (10.0f / 88.0f) : (10.0f / 32.0f);
    out[i] = (s * 2.0f - 1.0f) * scale;
  } else {
    out[i] = s;
  }
}

// ---------------------------------------------------------------- Gaussian top-25 sample + BEV scatter
__global__ void k_sample(const float* __restrict__ feats, const float* __restrict__ idxmap,
                         const int* __restrict__ bevidx, const float* __restrict__ mu,
                         const float* __restrict__ vp,
                         float* __restrict__ bev_feat, float* __restrict__ bev_conf) {
  __shared__ float sc[8][640];
  const int lane = threadIdx.x & 31;
  const int wave = threadIdx.x >> 5;
  const int n = blockIdx.x * 8 + wave;          // grid sized exactly: n < 16896
  const int b = n / P2B_HW, p = n % P2B_HW;

  float mux = mu[(b * 2 + 0) * P2B_HW + p];
  float muy = mu[(b * 2 + 1) * P2B_HW + p];
  float bx  = idxmap[(long)n * 2 + 0] * 2.0f - 1.0f;
  float by  = idxmap[(long)n * 2 + 1] * 2.0f - 1.0f;
  float meanx = fminf(fmaxf(bx + mux, -0.999f), 0.999f);
  float meany = fminf(fmaxf(by + muy, -0.999f), 0.999f);
  float v0 = vp[(b * 4 + 0) * P2B_HW + p];
  float v1 = vp[(b * 4 + 1) * P2B_HW + p];
  float v2 = vp[(b * 4 + 2) * P2B_HW + p];
  float conf = vp[(b * 4 + 3) * P2B_HW + p];
  const float r2 = (88.0f / 32.0f) * (88.0f / 32.0f);
  float vx = v0 * (P2B_MAXVAR - P2B_MINVAR) + P2B_MINVAR;
  float vy = v1 * (P2B_MAXVAR - P2B_MINVAR) * r2 + P2B_MINVAR * r2;
  float corr = v2 * 1.9f - 0.95f;
  float cov = corr * sqrtf(vx * vy);
  float det = vx * vy - cov * cov;
  float invdet = 1.0f / det;
  float xi = roundf((meanx + 1.0f) * 0.5f * 87.0f);
  float yi = roundf((meany + 1.0f) * 0.5f * 31.0f);
  int x_idx = (int)fminf(fmaxf(xi, 12.0f), 75.0f);
  int y_idx = (int)fminf(fmaxf(yi, 12.0f), 19.0f);

  float* s = sc[wave];
  for (int q = lane; q < 625; q += 32) {
    int i5 = q / 25, j5 = q % 25;
    float xc = (float)(x_idx + i5 - 12) * (2.0f / 87.0f) - 1.0f;
    float yc = (float)(y_idx + j5 - 12) * (2.0f / 31.0f) - 1.0f;
    float dx = xc - meanx, dy = yc - meany;
    float mahal = (vy * dx * dx - 2.0f * cov * dx * dy + vx * dy * dy) * invdet;
    s[q] = -0.5f * mahal;
  }
  __syncthreads();

  float selVal = 0.0f; int selIdx = 0;
  for (int t = 0; t < 25; ++t) {
    float bv = -1e30f; int bi = 0;
    for (int q = lane; q < 625; q += 32) {
      float v = s[q];
      if (v > bv) { bv = v; bi = q; }
    }
    for (int off = 16; off >= 1; off >>= 1) {
      float ov = __shfl_xor(bv, off, 32);
      int   oi = __shfl_xor(bi, off, 32);
      if (ov > bv || (ov == bv && oi < bi)) { bv = ov; bi = oi; }
    }
    if (lane == t) { selVal = bv; selIdx = bi; }
    if (lane == 0) s[bi] = -1e38f;
    __syncthreads();
  }

  float maxv = __shfl(selVal, 0, 32);
  float wv = (lane < 25) ? expf(selVal - maxv) : 0.0f;
  float sum = wv;
  for (int off = 16; off >= 1; off >>= 1) sum += __shfl_xor(sum, off, 32);
  float weight = wv / sum;

  float acc[8] = {0.f, 0.f, 0.f, 0.f, 0.f, 0.f, 0.f, 0.f};
  const float* fb = feats + (long)b * P2B_C * P2B_HW;
  for (int t = 0; t < 25; ++t) {
    float wt = __shfl(weight, t, 32);
    int idx  = __shfl(selIdx, t, 32);
    int gx = x_idx + idx / 25 - 12;
    int gy = y_idx + idx % 25 - 12;
    int lin = gy * P2B_WW + gx;
#pragma unroll
    for (int cc = 0; cc < 8; ++cc) {
      int c = lane + cc * 32;
      acc[cc] += wt * fb[(long)c * P2B_HW + lin];
    }
  }

  int seg = (b / P2B_NB) * P2B_BEVN + bevidx[n];
#pragma unroll
  for (int cc = 0; cc < 8; ++cc) {
    int c = lane + cc * 32;
    atomicAdd(&bev_feat[(long)seg * P2B_C + c], acc[cc] * conf);
  }
  if (lane == 0) atomicAdd(&bev_conf[seg], conf);
}

// ---------------------------------------------------------------- finalize
__global__ void k_final(const float* __restrict__ bev_feat, const float* __restrict__ bev_conf,
                        const float* __restrict__ fill, float* __restrict__ out) {
  long i = (long)blockIdx.x * blockDim.x + threadIdx.x;
  const long n = (long)P2B_C * P2B_BEVN;
  if (i >= n) return;
  int c  = (int)(i / P2B_BEVN);
  int hw = (int)(i % P2B_BEVN);
  float cf = fminf(fmaxf(bev_conf[hw], 0.0f), 1.0f);
  out[i] = bev_feat[(long)hw * P2B_C + c] + (1.0f - cf) * fill[c];
}

// ---------------------------------------------------------------- host launcher
extern "C" void kernel_launch(void* const* d_in, const int* in_sizes, int n_in,
                              void* d_out, int out_size, void* d_ws, size_t ws_size,
                              hipStream_t stream) {
  const float* feats  = (const float*)d_in[0];
  const float* mask   = (const float*)d_in[1];
  const float* idxmap = (const float*)d_in[2];
  const int*   bevidx = (const int*)d_in[3];
  const float* W1  = (const float*)d_in[4];
  const float* W2  = (const float*)d_in[5];
  const float* W3  = (const float*)d_in[6];
  const float* Wo1 = (const float*)d_in[7];
  const float* Wo2 = (const float*)d_in[8];
  const float* Wo3 = (const float*)d_in[9];
  const float* Wv1 = (const float*)d_in[10];
  const float* Wv2 = (const float*)d_in[11];
  const float* Wv3 = (const float*)d_in[12];
  const float* fill = (const float*)d_in[13];
  float* out = (float*)d_out;

  char* wsp = (char*)d_ws;
  auto carve = [&](size_t bytes) -> char* {
    char* p = wsp; wsp += (bytes + 255) & ~(size_t)255; return p;
  };
  const long PHW = P2B_PHW;                        // 3312 padded pixels per image
  const long n288 = (long)P2B_NB * PHW * 288;      // padded NHWC elements (288ch)
  const long n256 = (long)P2B_NB * PHW * 256;
  float*  t1       = (float*) carve((size_t)P2B_NB * 16 * P2B_HW * 4);
  float*  t2       = (float*) carve((size_t)P2B_NB * 32 * P2B_HW * 4);
  bf16_t* in1      = (bf16_t*)carve((size_t)n288 * 2);
  bf16_t* in2      = (bf16_t*)carve((size_t)n288 * 2);
  bf16_t* in3      = (bf16_t*)carve((size_t)n288 * 2);
  bf16_t* vbuf1    = (bf16_t*)carve((size_t)n256 * 2);
  bf16_t* vbuf2    = (bf16_t*)carve((size_t)n256 * 2);
  bf16_t* wo1p     = (bf16_t*)carve((size_t)256 * 288 * 9 * 2);
  bf16_t* wo2p     = (bf16_t*)carve((size_t)256 * 288 * 9 * 2);
  bf16_t* wv1p     = (bf16_t*)carve((size_t)256 * 256 * 9 * 2);
  bf16_t* wv2p     = (bf16_t*)carve((size_t)256 * 256 * 9 * 2);
  float*  mu       = (float*) carve((size_t)P2B_NB * 2 * P2B_HW * 4);
  float*  vp       = (float*) carve((size_t)P2B_NB * 4 * P2B_HW * 4);
  float*  bev_feat = (float*) carve((size_t)P2B_BEVN * P2B_C * 4);
  float*  bev_conf = (float*) carve((size_t)P2B_BEVN * 4);

  auto blk = [](long n) { return (int)((n + 255) / 256); };

  // zero padded activation buffers (halo must be 0) and BEV accumulators
  k_zero_f32<<<blk(n288 / 2), 256, 0, stream>>>((float*)in1, n288 / 2);
  k_zero_f32<<<blk(n288 / 2), 256, 0, stream>>>((float*)in2, n288 / 2);
  k_zero_f32<<<blk(n288 / 2), 256, 0, stream>>>((float*)in3, n288 / 2);
  k_zero_f32<<<blk(n256 / 2), 256, 0, stream>>>((float*)vbuf1, n256 / 2);
  k_zero_f32<<<blk(n256 / 2), 256, 0, stream>>>((float*)vbuf2, n256 / 2);
  k_zero_f32<<<blk((long)P2B_BEVN * P2B_C), 256, 0, stream>>>(bev_feat, (long)P2B_BEVN * P2B_C);
  k_zero_f32<<<blk(P2B_BEVN), 256, 0, stream>>>(bev_conf, P2B_BEVN);

  // pack weights into WMMA fragment order (bf16)
  k_pack_w<<<blk(16L * 9 * 9 * 512), 256, 0, stream>>>(Wo1, wo1p, 288, 16);
  k_pack_w<<<blk(16L * 9 * 9 * 512), 256, 0, stream>>>(Wo2, wo2p, 288, 16);
  k_pack_w<<<blk(16L * 9 * 8 * 512), 256, 0, stream>>>(Wv1, wv1p, 256, 16);
  k_pack_w<<<blk(16L * 9 * 8 * 512), 256, 0, stream>>>(Wv2, wv2p, 256, 16);

  // activations: feats -> padded NHWC bf16 (channels 0..255 of in1)
  k_feats_to_in<<<blk((long)P2B_NB * P2B_HW * P2B_C), 256, 0, stream>>>(feats, in1);

  // mask tower -> m into channels 256..287 of in1/in2/in3
  k_mask1<<<blk((long)P2B_NB * 16 * P2B_HW), 256, 0, stream>>>(mask, W1, t1);
  k_mask2<<<blk((long)P2B_NB * 32 * P2B_HW), 256, 0, stream>>>(t1, W2, t2);
  k_mask3<<<blk((long)P2B_NB * 32 * P2B_HW), 256, 0, stream>>>(t2, W3, in1, in2, in3);

  // big convs: 16 coutTiles * 88 double-pixel-tiles * 6 imgs = 8448 waves = 1056 blocks
  const int convBlocks = 16 * (P2B_HW / 32) * P2B_NB / 8;
  k_wmma_conv3x3<<<convBlocks, 256, 0, stream>>>(in1, 288, 288, wo1p, in2, 288, 16);
  k_wmma_conv3x3<<<convBlocks, 256, 0, stream>>>(in2, 288, 288, wo2p, in3, 288, 16);
  k_wmma_conv3x3<<<convBlocks, 256, 0, stream>>>(in1, 288, 256, wv1p, vbuf1, 256, 16);
  k_wmma_conv3x3<<<convBlocks, 256, 0, stream>>>(vbuf1, 256, 256, wv2p, vbuf2, 256, 16);

  // 5x5 heads
  k_head5<<<blk((long)P2B_NB * 2 * P2B_HW), 256, 0, stream>>>(in3, 288, 288, Wo3, 2, mu, 0);
  k_head5<<<blk((long)P2B_NB * 4 * P2B_HW), 256, 0, stream>>>(vbuf2, 256, 256, Wv3, 4, vp, 1);

  // Gaussian top-25 sampling + BEV scatter: one wave per pixel, 8 waves/block
  k_sample<<<(P2B_NB * P2B_HW) / 8, 256, 0, stream>>>(feats, idxmap, bevidx, mu, vp, bev_feat, bev_conf);

  // finalize into NCHW output
  k_final<<<blk((long)P2B_C * P2B_BEVN), 256, 0, stream>>>(bev_feat, bev_conf, fill, out);
}
